// InvariantDescriptorBuilder_79405355368870
// MI455X (gfx1250) — matrix operand order repaired and checked by reading, not compiled, not run on hardware
//
#include <hip/hip_runtime.h>

typedef __attribute__((ext_vector_type(16))) _Float16     v16h;
typedef __attribute__((ext_vector_type(8)))  float        v8f;
typedef __attribute__((ext_vector_type(8)))  unsigned int v8u;

#define NMAX      256          // j-dimension (K of the GEMM), padded/fixed
#define RNUM      8
#define WCOLS     112          // 7 tiles of 16 moment columns
#define OUT_PER_I 603

__device__ __forceinline__ v16h load_a_frag(const unsigned int* __restrict__ Arow,
                                            int kb, int hi) {
    // 16-bit A 16x32 layout: k = 2v + 8*(v>=4) + 8*hi  -> dword = v + 4*(v>=4) + 4*hi
    v8u u;
    const int ab = kb * 16 + 4 * hi;
    u[0] = Arow[ab + 0];  u[1] = Arow[ab + 1];
    u[2] = Arow[ab + 2];  u[3] = Arow[ab + 3];
    u[4] = Arow[ab + 8];  u[5] = Arow[ab + 9];
    u[6] = Arow[ab + 10]; u[7] = Arow[ab + 11];
    return __builtin_bit_cast(v16h, u);
}

__device__ __forceinline__ v16h load_b_frag(const unsigned int* __restrict__ Brow,
                                            int kb, int hi) {
    // 16-bit B 32x16 layout (matches A): k = 2v + 8*(v>=4) + 8*hi -> dword = v + 4*(v>=4) + 4*hi
    v8u u;
    const int bb = kb * 16 + 4 * hi;
#pragma unroll
    for (int v = 0; v < 8; ++v) u[v] = Brow[bb + v + 4 * (v >= 4)];
    return __builtin_bit_cast(v16h, u);
}

// Legendre-combined contraction of two moment rows (cols c0..c0+9):
// l0 = P0 term, l1 = P1 term, l2 = P2 term (before diagonal correction)
__device__ __forceinline__ void combine_moments(const float* __restrict__ Bm,
                                                int a, int b, int c0,
                                                float& l0, float& l1, float& l2) {
    const float* ra = Bm + a * WCOLS + c0;
    const float* rb = Bm + b * WCOLS + c0;
    l0 = ra[0] * rb[0];
    l1 = ra[1] * rb[1] + ra[2] * rb[2] + ra[3] * rb[3];
    float s2 = ra[4] * rb[4] + ra[5] * rb[5] + ra[6] * rb[6]
             + 2.0f * (ra[7] * rb[7] + ra[8] * rb[8] + ra[9] * rb[9]);
    l2 = 0.5f * (3.0f * s2 - l0);
}

__global__ __launch_bounds__(256) void
mag_invariant_descriptor_kernel(const float* __restrict__ pos,
                                const float* __restrict__ mag,
                                const float* __restrict__ cell,
                                float* __restrict__ out, int N) {
    constexpr float CUTOFF  = 4.5f;
    constexpr float SPACING = CUTOFF / 7.0f;               // linspace(0,4.5,8)
    constexpr float BETA    = 1.0f / (SPACING * SPACING);
    constexpr float U_REF   = 2.2f * 2.2f;
    constexpr float PI_F    = 3.14159265358979f;

    // Ft: A-operand, F^T as [radial a(16)][j(256)] f16  (8 KB) — reused as Bm (f32 16x112)
    // Wt: B-operand, W^T as [moment m(112)][j(256)] f16 (56 KB)
    __shared__ __align__(16) _Float16 Ft[16][NMAX];
    __shared__ __align__(16) _Float16 Wt[WCOLS][NMAX];

    const int i = blockIdx.x;
    const int t = threadIdx.x;

    if (t == 0) { __builtin_prefetch(pos, 0, 3); __builtin_prefetch(mag, 0, 3); }

    // ---- cell inverse (redundant per thread; 9 loads + ~30 flops) ----
    const float c00 = cell[0], c01 = cell[1], c02 = cell[2];
    const float c10 = cell[3], c11 = cell[4], c12 = cell[5];
    const float c20 = cell[6], c21 = cell[7], c22 = cell[8];
    const float det = c00 * (c11 * c22 - c12 * c21)
                    - c01 * (c10 * c22 - c12 * c20)
                    + c02 * (c10 * c21 - c11 * c20);
    const float id  = 1.0f / det;
    const float i00 = (c11 * c22 - c12 * c21) * id, i01 = (c02 * c21 - c01 * c22) * id,
                i02 = (c01 * c12 - c02 * c11) * id;
    const float i10 = (c12 * c20 - c10 * c22) * id, i11 = (c00 * c22 - c02 * c20) * id,
                i12 = (c02 * c10 - c00 * c12) * id;
    const float i20 = (c10 * c21 - c11 * c20) * id, i21 = (c01 * c20 - c00 * c21) * id,
                i22 = (c00 * c11 - c01 * c10) * id;

    const float pix = pos[3 * i + 0], piy = pos[3 * i + 1], piz = pos[3 * i + 2];
    const float mix = mag[3 * i + 0], miy = mag[3 * i + 1], miz = mag[3 * i + 2];

    // =============== Phase 1: build F (radial basis) and W (moment weights) ===============
    for (int j = t; j < NMAX; j += blockDim.x) {
        const bool valid = (j < N);
        const float pjx = valid ? pos[3 * j + 0] : 0.f;
        const float pjy = valid ? pos[3 * j + 1] : 0.f;
        const float pjz = valid ? pos[3 * j + 2] : 0.f;
        const float mjx = valid ? mag[3 * j + 0] : 0.f;
        const float mjy = valid ? mag[3 * j + 1] : 0.f;
        const float mjz = valid ? mag[3 * j + 2] : 0.f;

        // minimum-image displacement r_ij = wrap(pos_j - pos_i)
        const float dx = pjx - pix, dy = pjy - piy, dz = pjz - piz;
        float fx = dx * i00 + dy * i10 + dz * i20;
        float fy = dx * i01 + dy * i11 + dz * i21;
        float fz = dx * i02 + dy * i12 + dz * i22;
        fx -= rintf(fx); fy -= rintf(fy); fz -= rintf(fz);
        const float rx = fx * c00 + fy * c10 + fz * c20;
        const float ry = fx * c01 + fy * c11 + fz * c21;
        const float rz = fx * c02 + fy * c12 + fz * c22;
        const float d  = sqrtf(rx * rx + ry * ry + rz * rz);

        const bool nbr = valid && (d < CUTOFF) && (j != i);
        const float fc = nbr ? 0.5f * (__cosf(PI_F * d / CUTOFF) + 1.0f) : 0.0f;
        const float invd = (d > 1e-12f) ? 1.0f / d : 0.0f;
        const float ux = rx * invd, uy = ry * invd, uz = rz * invd;

        const float uj = mjx * mjx + mjy * mjy + mjz * mjz;   // |m_j|^2
        const float s  = mix * mjx + miy * mjy + miz * mjz;   // m_i . m_j

        float f[RNUM];
#pragma unroll
        for (int a = 0; a < RNUM; ++a) {
            const float diff = d - (float)a * SPACING;
            f[a] = fc * __expf(-BETA * diff * diff);
        }
#pragma unroll
        for (int a = 0; a < RNUM; ++a) Ft[a][j] = (_Float16)f[a];
#pragma unroll
        for (int a = RNUM; a < 16; ++a) Ft[a][j] = (_Float16)0.f;

        // moment weights: {1, ux,uy,uz, ux^2,uy^2,uz^2, ux*uy,ux*uz,uy*uz}
        float w[10];
        w[0] = 1.0f;
        w[1] = ux; w[2] = uy; w[3] = uz;
        w[4] = ux * ux; w[5] = uy * uy; w[6] = uz * uz;
        w[7] = ux * uy; w[8] = ux * uz; w[9] = uy * uz;

        // tile 0: plain moments + u_j
#pragma unroll
        for (int m = 0; m < 10; ++m) Wt[m][j] = (_Float16)w[m];
        Wt[10][j] = (_Float16)uj;
#pragma unroll
        for (int m = 11; m < 16; ++m) Wt[m][j] = (_Float16)0.f;
        // tile 1: moments * s_ij
#pragma unroll
        for (int m = 0; m < 10; ++m) Wt[16 + m][j] = (_Float16)(w[m] * s);
#pragma unroll
        for (int m = 26; m < 32; ++m) Wt[m][j] = (_Float16)0.f;
        // tiles 2..4: moments * m_j{x,y,z}
#pragma unroll
        for (int m = 0; m < 10; ++m) Wt[32 + m][j] = (_Float16)(w[m] * mjx);
#pragma unroll
        for (int m = 42; m < 48; ++m) Wt[m][j] = (_Float16)0.f;
#pragma unroll
        for (int m = 0; m < 10; ++m) Wt[48 + m][j] = (_Float16)(w[m] * mjy);
#pragma unroll
        for (int m = 58; m < 64; ++m) Wt[m][j] = (_Float16)0.f;
#pragma unroll
        for (int m = 0; m < 10; ++m) Wt[64 + m][j] = (_Float16)(w[m] * mjz);
#pragma unroll
        for (int m = 74; m < 80; ++m) Wt[m][j] = (_Float16)0.f;
        // tile 5: f_b and f_b*u_j  (diagonal corrections for a_rr / a_mm)
#pragma unroll
        for (int b = 0; b < RNUM; ++b) Wt[80 + b][j] = (_Float16)f[b];
#pragma unroll
        for (int b = 0; b < RNUM; ++b) Wt[88 + b][j] = (_Float16)(f[b] * uj);
        // tile 6: f_b*s^2 (diagonal for a_imm) + pad
#pragma unroll
        for (int b = 0; b < RNUM; ++b) Wt[96 + b][j] = (_Float16)(f[b] * s * s);
#pragma unroll
        for (int m = 104; m < 112; ++m) Wt[m][j] = (_Float16)0.f;
    }
    __syncthreads();

    // =============== Phase 2: Bm = F^T (16x256) x W (256x112) via WMMA ===============
    const int lane = t & 31;
    const int wv   = t >> 5;           // 8 wave32's; waves 0..6 own one 16-col tile each
    const int m16  = lane & 15;
    const int hi   = lane >> 4;

    v8f acc = {0.f, 0.f, 0.f, 0.f, 0.f, 0.f, 0.f, 0.f};
    if (wv < 7) {
        const unsigned int* Arow = (const unsigned int*)&Ft[m16][0];
        const unsigned int* Brow = (const unsigned int*)&Wt[wv * 16 + m16][0];
#pragma unroll
        for (int kb = 0; kb < NMAX / 32; ++kb) {
            const v16h af = load_a_frag(Arow, kb, hi);
            const v16h bf = load_b_frag(Brow, kb, hi);
            acc = __builtin_amdgcn_wmma_f32_16x16x32_f16(
                      false, af, false, bf, (short)0, acc, false, false);
        }
    }
    __syncthreads();                    // all waves done reading Ft before overlaying Bm

    float* Bm = reinterpret_cast<float*>(&Ft[0][0]);  // 16x112 f32 = 7 KB, fits in Ft
    if (wv < 7) {
#pragma unroll
        for (int v = 0; v < 8; ++v)
            Bm[(v + 8 * hi) * WCOLS + wv * 16 + m16] = acc[v];
    }
    __syncthreads();

    // =============== Phase 3: combine moments -> 603 descriptors ===============
    float* out_i = out + (size_t)i * OUT_PER_I;

    if (t < 64) {
        const int a = t >> 3, b = t & 7;
        const float D0 = Bm[a * WCOLS + 80 + b];  // sum f_a f_b           (j==k, P_l(1)=1)
        const float Dm = Bm[a * WCOLS + 88 + b];  // sum f_a f_b |m_j|^2
        const float Di = Bm[a * WCOLS + 96 + b];  // sum f_a f_b s_ij^2
        float l0, l1, l2;

        // a_rr
        combine_moments(Bm, a, b, 0, l0, l1, l2);
        {
            const int base = 11 + a * 24 + b * 3;
            out_i[base + 0] = l0 - D0;
            out_i[base + 1] = l1 - D0;
            out_i[base + 2] = l2 - D0;
        }
        // a_mm: sum over magnetization components e in {x,y,z}
        {
            float m0 = 0.f, m1 = 0.f, m2 = 0.f;
#pragma unroll
            for (int e = 0; e < 3; ++e) {
                combine_moments(Bm, a, b, 32 + 16 * e, l0, l1, l2);
                m0 += l0; m1 += l1; m2 += l2;
            }
            const int base = 219 + a * 24 + b * 3;
            out_i[base + 0] = m0 - Dm;
            out_i[base + 1] = m1 - Dm;
            out_i[base + 2] = m2 - Dm;
        }
        // a_imm
        combine_moments(Bm, a, b, 16, l0, l1, l2);
        {
            const int base = 411 + a * 24 + b * 3;
            out_i[base + 0] = l0 - Di;
            out_i[base + 1] = l1 - Di;
            out_i[base + 2] = l2 - Di;
        }
    } else if (t < 72) {
        const int a = t - 64;
        out_i[3 + a]   = Bm[a * WCOLS + 0];    // rho_r
        out_i[203 + a] = Bm[a * WCOLS + 10];   // rho_m2 (f * u_j)
        out_i[211 + a] = Bm[a * WCOLS + 16];   // rho_im (f * s_ij)
    } else if (t == 72) {
        const float un = (mix * mix + miy * miy + miz * miz) / U_REF;
        out_i[0] = un;
        out_i[1] = un * un;
        out_i[2] = un * un * un;
    }
}

extern "C" void kernel_launch(void* const* d_in, const int* in_sizes, int n_in,
                              void* d_out, int out_size, void* d_ws, size_t ws_size,
                              hipStream_t stream) {
    const float* pos  = (const float*)d_in[0];
    const float* mag  = (const float*)d_in[1];
    const float* cell = (const float*)d_in[2];
    float* out = (float*)d_out;
    const int N = in_sizes[0] / 3;   // 256
    mag_invariant_descriptor_kernel<<<N, 256, 0, stream>>>(pos, mag, cell, out, N);
}
